// Generator_103079215776
// MI455X (gfx1250) — matrix operand order repaired and checked by reading, not compiled, not run on hardware
//
#include <hip/hip_runtime.h>

typedef __attribute__((ext_vector_type(16))) _Float16 v16h;
typedef __attribute__((ext_vector_type(8)))  _Float16 v8h;
typedef __attribute__((ext_vector_type(8)))  float    v8f;

#define WPB 4  // waves per block (wave32)
#define MB  2  // M-tiles per wave (rows:  MB*16)
#define NB  2  // N-tiles per wave (chans: NB*16)

static __device__ __forceinline__ float frelu(float v){ return v > 0.f ? v : 0.f; }

static __device__ __forceinline__ v16h ld_frag(const _Float16* p0, const _Float16* p1){
  v8h a0 = *(const v8h*)p0;
  v8h a1 = *(const v8h*)p1;
  return __builtin_shufflevector(a0, a1, 0,1,2,3,4,5,6,7,8,9,10,11,12,13,14,15);
}

// ---------------------------------------------------------------- utility
__global__ void k_zero(float* p, int n){
  for (int i = blockIdx.x*blockDim.x + threadIdx.x; i < n; i += gridDim.x*blockDim.x) p[i] = 0.f;
}

// x NCHW f32 -> a NHWC f16 with channel padding to Cp (zeros for c >= C)
__global__ void k_x_to_nhwc(const float* __restrict__ x, _Float16* __restrict__ a,
                            int C, int Cp, int H, int W, long long total){
  for (long long i = blockIdx.x*(long long)blockDim.x + threadIdx.x; i < total;
       i += (long long)gridDim.x*blockDim.x){
    int c = (int)(i % Cp); long long t = i / Cp;
    int w = (int)(t % W); t /= W;
    int h = (int)(t % H); int n = (int)(t / H);
    a[i] = (c < C) ? (_Float16)x[(((size_t)n*C + c)*H + h)*W + w] : (_Float16)0.0f;
  }
}

__global__ void k_nhwc_to_nchw(const float* __restrict__ y, float* __restrict__ out,
                               int C, int H, int W, long long total){
  for (long long i = blockIdx.x*(long long)blockDim.x + threadIdx.x; i < total;
       i += (long long)gridDim.x*blockDim.x){
    // i indexes NCHW output
    int w = (int)(i % W); long long t = i / W;
    int h = (int)(t % H); t /= H;
    int c = (int)(t % C); int n = (int)(t / C);
    out[i] = y[(((size_t)n*H + h)*W + w)*C + c];
  }
}

__global__ void k_f32_to_f16(const float* __restrict__ y, _Float16* __restrict__ a, long long total){
  for (long long i = blockIdx.x*(long long)blockDim.x + threadIdx.x; i < total;
       i += (long long)gridDim.x*blockDim.x)
    a[i] = (_Float16)y[i];
}

// weight transform -> f16, layout [kh*k+kw][Coutp][Cinp], zero-padded
// mode 0: w is [O,I,k,k] standard conv
// mode 1: w is [I,O,k,k] (ConvTranspose); effective conv kernel is flipped+transposed
__global__ void k_wt(const float* __restrict__ w, _Float16* __restrict__ wt,
                     int O, int I, int Op, int Ip, int ksz, int mode, long long total){
  for (long long i = blockIdx.x*(long long)blockDim.x + threadIdx.x; i < total;
       i += (long long)gridDim.x*blockDim.x){
    int ii = (int)(i % Ip); long long t = i / Ip;
    int o = (int)(t % Op); int khw = (int)(t / Op);
    int kh = khw / ksz, kw = khw % ksz;
    float v = 0.f;
    if (o < O && ii < I){
      if (mode == 0) v = w[(((size_t)o*I + ii)*ksz + kh)*ksz + kw];
      else {
        int fh = ksz-1-kh, fw = ksz-1-kw;
        v = w[(((size_t)ii*O + o)*ksz + fh)*ksz + fw];
      }
    }
    wt[i] = (_Float16)v;
  }
}

__global__ void k_wsq(const float* __restrict__ cb, float* __restrict__ wsq, int K, int D){
  int k = blockIdx.x*blockDim.x + threadIdx.x;
  if (k < K){
    float s = 0.f;
    const float* row = cb + (size_t)k*D;
    for (int d = 0; d < D; ++d) s += row[d]*row[d];
    wsq[k] = s;
  }
}

// ---------------------------------------------------------------- WMMA implicit-GEMM conv
// act: f16 NHWC [N*Hin*Win, Cinp]; wt: f16 [tap][Coutp][Cinp]; y: f32 [N*Hout*Wout, Cout]
// zp: zero page (>= Cinp f16 zeros) for invalid spatial taps -> branch-free loads
// Register-blocked MB x NB: each wave computes (MB*16) rows x (NB*16) channels.
// mode 0: normal conv (ih = oh*stride - pad + kh)
// mode 1: input-dilated (transposed) conv: th = oh + pad - kh, valid iff th%stride==0
__global__ void k_conv(const _Float16* __restrict__ act, const _Float16* __restrict__ wt,
                       const float* __restrict__ bias, const _Float16* __restrict__ zp,
                       float* __restrict__ y,
                       int Hin, int Win, int Cinp, int Hout, int Wout, int Cout, int Coutp,
                       int ksz, int stride, int pad, int mode, int Mt2){
  int wave = threadIdx.x >> 5, lane = threadIdx.x & 31;
  int mtile = blockIdx.y * WPB + wave;      // index over MB-row-tile groups
  if (mtile >= Mt2) return;
  int nn = lane & 15;
  int hs = lane >> 4;                       // half-select (lanes 16-31)
  int HW = Hout*Wout;

  int n_[MB], oh_[MB], ow_[MB];
  #pragma unroll
  for (int mb = 0; mb < MB; ++mb){
    int rowA = (mtile*MB + mb)*16 + nn;     // A-matrix row this lane loads
    n_[mb] = rowA / HW; int rem = rowA % HW;
    oh_[mb] = rem / Wout; ow_[mb] = rem % Wout;
  }
  int o_[NB];
  #pragma unroll
  for (int nb = 0; nb < NB; ++nb) o_[nb] = (blockIdx.x*NB + nb)*16 + nn;  // < Coutp (padded)

  int ncc = Cinp >> 5;
  v8f acc[MB][NB];
  #pragma unroll
  for (int mb = 0; mb < MB; ++mb)
    #pragma unroll
    for (int nb = 0; nb < NB; ++nb) acc[mb][nb] = (v8f){};

  for (int kh = 0; kh < ksz; ++kh){
    for (int kw = 0; kw < ksz; ++kw){
      const _Float16* ap[MB];
      #pragma unroll
      for (int mb = 0; mb < MB; ++mb){
        int ih, iw; bool vh, vw;
        if (mode == 0){
          ih = oh_[mb]*stride - pad + kh; vh = (ih >= 0) && (ih < Hin);
          iw = ow_[mb]*stride - pad + kw; vw = (iw >= 0) && (iw < Win);
        } else {
          int th = oh_[mb] + pad - kh; vh = (th >= 0) && (th % stride == 0); ih = th / stride; vh = vh && (ih < Hin);
          int tw = ow_[mb] + pad - kw; vw = (tw >= 0) && (tw % stride == 0); iw = tw / stride; vw = vw && (iw < Win);
        }
        ap[mb] = (vh && vw) ? (act + (((size_t)n_[mb]*Hin + ih)*Win + iw) * (size_t)Cinp) : zp;
      }
      const _Float16* wp[NB];
      #pragma unroll
      for (int nb = 0; nb < NB; ++nb)
        wp[nb] = wt + ((size_t)(kh*ksz + kw)*Coutp + o_[nb]) * (size_t)Cinp;

      for (int cc = 0; cc < ncc; ++cc){
        int c0 = cc << 5;
        v16h af[MB], bf[NB];
        #pragma unroll
        for (int mb = 0; mb < MB; ++mb)
          af[mb] = ld_frag(ap[mb] + c0 + hs*8, ap[mb] + c0 + 16 + hs*8);   // A 16x32 layout
        #pragma unroll
        for (int nb = 0; nb < NB; ++nb)
          bf[nb] = ld_frag(wp[nb] + c0 + hs*8, wp[nb] + c0 + 16 + hs*8);  // B 32x16 layout
        #pragma unroll
        for (int mb = 0; mb < MB; ++mb)
          #pragma unroll
          for (int nb = 0; nb < NB; ++nb)
            acc[mb][nb] = __builtin_amdgcn_wmma_f32_16x16x32_f16(false, af[mb], false, bf[nb],
                                                                 (short)0, acc[mb][nb], false, false);
      }
    }
  }
  #pragma unroll
  for (int nb = 0; nb < NB; ++nb){
    if (o_[nb] < Cout){
      float bv = bias[o_[nb]];
      #pragma unroll
      for (int mb = 0; mb < MB; ++mb){
        #pragma unroll
        for (int r = 0; r < 8; ++r){
          int m = (mtile*MB + mb)*16 + r + hs*8;          // C/D f32 lane layout
          y[(size_t)m*Cout + o_[nb]] = acc[mb][nb][r] + bv;
        }
      }
    }
  }
}

// ---------------------------------------------------------------- BN
__global__ void k_bnstats(const float* __restrict__ y, float* __restrict__ sum,
                          float* __restrict__ sq, long long total, int C){
  __shared__ float ls[256], lq[256];
  if ((int)threadIdx.x < 256){ ls[threadIdx.x] = 0.f; lq[threadIdx.x] = 0.f; }
  __syncthreads();
  for (long long i = blockIdx.x*(long long)blockDim.x + threadIdx.x; i < total;
       i += (long long)gridDim.x*blockDim.x){
    float v = y[i];
    int c = (int)(i % C);
    atomicAdd(&ls[c], v);
    atomicAdd(&lq[c], v*v);
  }
  __syncthreads();
  if ((int)threadIdx.x < C){
    atomicAdd(&sum[threadIdx.x], ls[threadIdx.x]);
    atomicAdd(&sq[threadIdx.x],  lq[threadIdx.x]);
  }
}

__global__ void k_bnrelu(const float* __restrict__ y, const float* __restrict__ sum,
                         const float* __restrict__ sq, const float* __restrict__ g,
                         const float* __restrict__ b, _Float16* __restrict__ out,
                         long long total, int C, float invcnt){
  for (long long i = blockIdx.x*(long long)blockDim.x + threadIdx.x; i < total;
       i += (long long)gridDim.x*blockDim.x){
    int c = (int)(i % C);
    float mean = sum[c]*invcnt;
    float var  = sq[c]*invcnt - mean*mean;
    float sc   = g[c]*rsqrtf(var + 1e-5f);
    float sh   = b[c] - mean*sc;
    out[i] = (_Float16)frelu(y[i]*sc + sh);
  }
}

// ---------------------------------------------------------------- VQ: WMMA distance + argmin
// D fixed at 256: the whole Z row is hoisted into registers (8 fragments) and
// reused across all 16 code tiles -> inner loop is 2 B-loads + 1 WMMA.
__global__ void k_vq(const _Float16* __restrict__ z, const _Float16* __restrict__ cbh,
                     const float* __restrict__ wsq, int* __restrict__ jout,
                     float* __restrict__ mv, int K, int Mtiles){
  const int D = 256;
  int wave = threadIdx.x >> 5, lane = threadIdx.x & 31;
  int mtile = blockIdx.x * WPB + wave;
  if (mtile >= Mtiles) return;
  int nn = lane & 15, hs = lane >> 4;
  const _Float16* zr = z + (size_t)(mtile*16 + nn) * D;

  v16h afr[8];
  #pragma unroll
  for (int cc = 0; cc < 8; ++cc){
    int c0 = cc << 5;
    afr[cc] = ld_frag(zr + c0 + hs*8, zr + c0 + 16 + hs*8);
  }

  float bval[8]; int bidx[8];
  #pragma unroll
  for (int r = 0; r < 8; ++r){ bval[r] = 3.402823466e38f; bidx[r] = 0x7fffffff; }

  for (int kt = 0; kt < (K >> 4); ++kt){
    int code = kt*16 + nn;
    const _Float16* cr = cbh + (size_t)code * D;
    v8f acc = {};
    #pragma unroll
    for (int cc = 0; cc < 8; ++cc){
      int c0 = cc << 5;
      v16h bf = ld_frag(cr + c0 + hs*8, cr + c0 + 16 + hs*8);
      acc = __builtin_amdgcn_wmma_f32_16x16x32_f16(false, afr[cc], false, bf,
                                                   (short)0, acc, false, false);
    }
    float wq = wsq[code];
    #pragma unroll
    for (int r = 0; r < 8; ++r){
      float sv = wq - 2.0f*acc[r];   // d2 minus per-row ||z||^2 (argmin-invariant)
      int   si = code;
      #pragma unroll
      for (int off = 8; off >= 1; off >>= 1){
        float ov = __shfl_xor(sv, off, 16);
        int   oi = __shfl_xor(si, off, 16);
        if (ov < sv || (ov == sv && oi < si)){ sv = ov; si = oi; }
      }
      if (sv < bval[r] || (sv == bval[r] && si < bidx[r])){ bval[r] = sv; bidx[r] = si; }
    }
  }
  if (nn == 0){
    #pragma unroll
    for (int r = 0; r < 8; ++r){
      int m = mtile*16 + r + hs*8;
      jout[m] = bidx[r];
      mv[m]   = bval[r];
    }
  }
}

__global__ void k_loss(const float* __restrict__ Z, const float* __restrict__ mv,
                       float* __restrict__ acc, int M, int D){
  float s = 0.f;
  int m = blockIdx.x*blockDim.x + threadIdx.x;
  if (m < M){
    const float* z = Z + (size_t)m*D;
    for (int d = 0; d < D; ++d) s += z[d]*z[d];
    s += mv[m];
  }
  __shared__ float red[256];
  red[threadIdx.x] = s;
  __syncthreads();
  for (int o = 128; o > 0; o >>= 1){
    if ((int)threadIdx.x < o) red[threadIdx.x] += red[threadIdx.x + o];
    __syncthreads();
  }
  if (threadIdx.x == 0) atomicAdd(acc, red[0]);
}

__global__ void k_gather(const float* __restrict__ cb, const int* __restrict__ j,
                         _Float16* __restrict__ hq, long long total, int D){
  for (long long i = blockIdx.x*(long long)blockDim.x + threadIdx.x; i < total;
       i += (long long)gridDim.x*blockDim.x){
    int m = (int)(i / D), c = (int)(i % D);
    hq[i] = (_Float16)cb[(size_t)j[m]*D + c];
  }
}

__global__ void k_finalloss(const float* __restrict__ acc, float* __restrict__ out, float invM){
  if (threadIdx.x == 0 && blockIdx.x == 0){
    float v = acc[0]*invM;
    out[0] = v;      // loss_commit
    out[1] = v;      // loss_code (identical value; stop_gradient only affects grads)
  }
}

// ---------------------------------------------------------------- host
static inline int nblk(long long n){
  long long b = (n + 255) >> 8;
  if (b < 1) b = 1;
  if (b > 8192) b = 8192;
  return (int)b;
}

extern "C" void kernel_launch(void* const* d_in, const int* in_sizes, int n_in,
                              void* d_out, int out_size, void* d_ws, size_t ws_size,
                              hipStream_t stream){
  (void)in_sizes; (void)n_in; (void)out_size; (void)ws_size;
  const float* x    = (const float*)d_in[0];
  const float* cb   = (const float*)d_in[1];
  const float* e_w1 = (const float*)d_in[2];  const float* e_b1 = (const float*)d_in[3];
  const float* e_g1 = (const float*)d_in[4];  const float* e_be1= (const float*)d_in[5];
  const float* e_w2 = (const float*)d_in[6];  const float* e_b2 = (const float*)d_in[7];
  const float* e_g2 = (const float*)d_in[8];  const float* e_be2= (const float*)d_in[9];
  const float* e_w3 = (const float*)d_in[10]; const float* e_b3 = (const float*)d_in[11];
  const float* e_g3 = (const float*)d_in[12]; const float* e_be3= (const float*)d_in[13];
  const float* e_w4 = (const float*)d_in[14]; const float* e_b4 = (const float*)d_in[15];
  const float* d_w1 = (const float*)d_in[16]; const float* d_b1 = (const float*)d_in[17];
  const float* d_g1 = (const float*)d_in[18]; const float* d_be1= (const float*)d_in[19];
  const float* d_w2 = (const float*)d_in[20]; const float* d_b2 = (const float*)d_in[21];
  const float* d_g2 = (const float*)d_in[22]; const float* d_be2= (const float*)d_in[23];
  const float* d_w3 = (const float*)d_in[24]; const float* d_b3 = (const float*)d_in[25];
  const float* d_g3 = (const float*)d_in[26]; const float* d_be3= (const float*)d_in[27];
  const float* d_w4 = (const float*)d_in[28]; const float* d_b4 = (const float*)d_in[29];
  float* gout = (float*)d_out;

  // -------- workspace carve
  char* ws = (char*)d_ws;
  size_t off = 0;
  auto carve = [&](size_t bytes)->void*{
    void* p = ws + off;
    off = (off + bytes + 255) & ~(size_t)255;
    return p;
  };
  _Float16* zp   = (_Float16*)carve(1024);                // zero page (>= max Cinp halfs)
  _Float16* wtE1 = (_Float16*)carve((size_t)9*64*32*2);   // [tap][Coutp][Cinp]
  _Float16* wtE2 = (_Float16*)carve((size_t)16*128*64*2);
  _Float16* wtE3 = (_Float16*)carve((size_t)16*192*128*2);
  _Float16* wtE4 = (_Float16*)carve((size_t)256*192*2);
  _Float16* wtD1 = (_Float16*)carve((size_t)192*256*2);
  _Float16* wtD2 = (_Float16*)carve((size_t)16*128*192*2);
  _Float16* wtD3 = (_Float16*)carve((size_t)16*64*128*2);
  _Float16* wtD4 = (_Float16*)carve((size_t)9*32*64*2);   // Cout 3 -> padded 32
  _Float16* cb16 = (_Float16*)carve((size_t)65536*2);
  float*    wsq  = (float*)carve(256*4);
  float*    stats= (float*)carve(4096*4);            // 6 layers * (sum256+sq256) + loss acc @ [3072]
  int*      jbuf = (int*)carve((size_t)32768*4);
  float*    minv = (float*)carve((size_t)32768*4);
  _Float16* act  = (_Float16*)carve((size_t)33554432*2);  // max activation (f16 NHWC)
  float*    ybuf = (float*)carve((size_t)33554432*4);     // max conv output (f32 NHWC)
  float* lossAcc = stats + 3072;

  auto SUM = [&](int L){ return stats + (size_t)L*512; };
  auto SQ  = [&](int L){ return stats + (size_t)L*512 + 256; };

  auto conv = [&](const _Float16* a, const _Float16* w, const float* bias, float* y,
                  int Hin, int Win, int Cinp, int Hout, int Wout, int Cout,
                  int ksz, int stride, int pad, int mode){
    int Coutp = (Cout + 31) & ~31;         // multiple of 32 = 16*NB
    int M = 128*Hout*Wout;
    int Mt = M/16;                         // always even -> MB=2 needs no guard
    int Mt2 = Mt/MB;
    dim3 grid(Coutp/(16*NB), (Mt2 + WPB - 1)/WPB);
    k_conv<<<grid, 32*WPB, 0, stream>>>(a, w, bias, zp, y, Hin, Win, Cinp,
                                        Hout, Wout, Cout, Coutp, ksz, stride, pad, mode, Mt2);
  };

  // -------- init
  k_zero<<<16, 256, 0, stream>>>(stats, 4096);
  k_zero<<<1, 256, 0, stream>>>((float*)zp, 256);
  {
    long long t = (long long)128*64*64*32;   // padded NHWC input
    k_x_to_nhwc<<<nblk(t), 256, 0, stream>>>(x, act, 3, 32, 64, 64, t);
  }
  k_wt<<<nblk(9*64*32),    256, 0, stream>>>(e_w1, wtE1, 64,  3,   64,  32,  3, 0, 9*64*32);
  k_wt<<<nblk(16*128*64),  256, 0, stream>>>(e_w2, wtE2, 128, 64,  128, 64,  4, 0, 16*128*64);
  k_wt<<<nblk(16*192*128), 256, 0, stream>>>(e_w3, wtE3, 192, 128, 192, 128, 4, 0, 16*192*128);
  k_wt<<<nblk(256*192),    256, 0, stream>>>(e_w4, wtE4, 256, 192, 256, 192, 1, 0, 256*192);
  k_wt<<<nblk(192*256),    256, 0, stream>>>(d_w1, wtD1, 192, 256, 192, 256, 1, 1, 192*256);
  k_wt<<<nblk(16*128*192), 256, 0, stream>>>(d_w2, wtD2, 128, 192, 128, 192, 4, 1, 16*128*192);
  k_wt<<<nblk(16*64*128),  256, 0, stream>>>(d_w3, wtD3, 64,  128, 64,  128, 4, 1, 16*64*128);
  k_wt<<<nblk(9*32*64),    256, 0, stream>>>(d_w4, wtD4, 3,   64,  32,  64,  3, 0, 9*32*64);
  k_f32_to_f16<<<nblk(65536), 256, 0, stream>>>(cb, cb16, 65536);
  k_wsq<<<1, 256, 0, stream>>>(cb, wsq, 256, 256);

  // -------- encoder
  conv(act, wtE1, e_b1, ybuf, 64, 64, 32,  64, 64, 64,  3, 1, 1, 0);
  { long long t = 33554432LL;
    k_bnstats<<<2048, 256, 0, stream>>>(ybuf, SUM(0), SQ(0), t, 64);
    k_bnrelu <<<nblk(t), 256, 0, stream>>>(ybuf, SUM(0), SQ(0), e_g1, e_be1, act, t, 64, 64.f/(float)t); }
  conv(act, wtE2, e_b2, ybuf, 64, 64, 64,  32, 32, 128, 4, 2, 1, 0);
  { long long t = 16777216LL;
    k_bnstats<<<2048, 256, 0, stream>>>(ybuf, SUM(1), SQ(1), t, 128);
    k_bnrelu <<<nblk(t), 256, 0, stream>>>(ybuf, SUM(1), SQ(1), e_g2, e_be2, act, t, 128, 128.f/(float)t); }
  conv(act, wtE3, e_b3, ybuf, 32, 32, 128, 16, 16, 192, 4, 2, 1, 0);
  { long long t = 6291456LL;
    k_bnstats<<<2048, 256, 0, stream>>>(ybuf, SUM(2), SQ(2), t, 192);
    k_bnrelu <<<nblk(t), 256, 0, stream>>>(ybuf, SUM(2), SQ(2), e_g3, e_be3, act, t, 192, 192.f/(float)t); }
  conv(act, wtE4, e_b4, ybuf, 16, 16, 192, 16, 16, 256, 1, 1, 0, 0);   // ybuf = Z [32768,256]

  // -------- vector quantization
  k_f32_to_f16<<<nblk(8388608LL), 256, 0, stream>>>(ybuf, act, 8388608LL);   // z16
  { int Mt = 32768/16;
    k_vq<<<(Mt + WPB - 1)/WPB, 32*WPB, 0, stream>>>(act, cb16, wsq, jbuf, minv, 256, Mt); }
  k_loss<<<128, 256, 0, stream>>>(ybuf, minv, lossAcc, 32768, 256);
  k_gather<<<nblk(8388608LL), 256, 0, stream>>>(cb, jbuf, act, 8388608LL, 256); // hq16

  // -------- decoder (transposed convs as input-dilated convs; pad arg = k-1-p)
  conv(act, wtD1, d_b1, ybuf, 16, 16, 256, 16, 16, 192, 1, 1, 0, 1);
  { long long t = 6291456LL;
    k_bnstats<<<2048, 256, 0, stream>>>(ybuf, SUM(3), SQ(3), t, 192);
    k_bnrelu <<<nblk(t), 256, 0, stream>>>(ybuf, SUM(3), SQ(3), d_g1, d_be1, act, t, 192, 192.f/(float)t); }
  conv(act, wtD2, d_b2, ybuf, 16, 16, 192, 32, 32, 128, 4, 2, 2, 1);
  { long long t = 16777216LL;
    k_bnstats<<<2048, 256, 0, stream>>>(ybuf, SUM(4), SQ(4), t, 128);
    k_bnrelu <<<nblk(t), 256, 0, stream>>>(ybuf, SUM(4), SQ(4), d_g2, d_be2, act, t, 128, 128.f/(float)t); }
  conv(act, wtD3, d_b3, ybuf, 32, 32, 128, 64, 64, 64,  4, 2, 2, 1);
  { long long t = 33554432LL;
    k_bnstats<<<2048, 256, 0, stream>>>(ybuf, SUM(5), SQ(5), t, 64);
    k_bnrelu <<<nblk(t), 256, 0, stream>>>(ybuf, SUM(5), SQ(5), d_g3, d_be3, act, t, 64, 64.f/(float)t); }
  conv(act, wtD4, d_b4, ybuf, 64, 64, 64,  64, 64, 3,   3, 1, 1, 0);

  // -------- outputs
  { long long t = (long long)128*3*64*64;
    k_nhwc_to_nchw<<<nblk(t), 256, 0, stream>>>(ybuf, gout, 3, 64, 64, t); }
  k_finalloss<<<1, 32, 0, stream>>>(lossAcc, gout + 1572864, 1.f/32768.f);
}
